// DCTFFN_2765958939260
// MI455X (gfx1250) — compile-verified, matrix operand order checked by reading, not run on hardware
//
#include <hip/hip_runtime.h>
#include <hip/hip_bf16.h>
#include <math.h>

typedef __attribute__((ext_vector_type(16))) _Float16 v16h;
typedef __attribute__((ext_vector_type(8)))  float    v8f;

#define C2    340      // 2*hidden
#define HID   170
#define CIN   64
#define IMG   256
#define IMG2  65536    // 256*256

union HPack { _Float16 h[2]; unsigned u; };

// ---------------------------------------------------------------------------
// Kernel 1: project_in GEMM (f16 WMMA) + fused 8x8 DCT -> quant -> IDCT.
// Block = 16 output channels x (8 rows x 32 cols) tile. Writes yr as fp16.
// ---------------------------------------------------------------------------
__global__ __launch_bounds__(256) void dctffn_k1(
    const float* __restrict__ x, const float* __restrict__ W_in,
    const float* __restrict__ quant, _Float16* __restrict__ yr)
{
  // LDS carve-up (51456 B):
  //   [0)      lds_m   float[64]            256 B   DCT basis
  //   [256)    lds_w   f16[16*64]          2048 B   A fragments (W_in chunk)
  //   [2304)   lds_x   f16[256*64]        32768 B   B fragments (x tile, pixel-major)
  //            lds_t   float (aliases lds_x after GEMM; needs 16384 B)
  //   [35072)  lds_y   float[16*256]      16384 B   y tile / IDCT result
  __shared__ __align__(32) unsigned char smem[51456];
  float*    lds_m = (float*)smem;
  _Float16* lds_w = (_Float16*)(smem + 256);
  _Float16* lds_x = (_Float16*)(smem + 2304);
  float*    lds_t = (float*)(smem + 2304);
  float*    lds_y = (float*)(smem + 35072);

  const int t    = threadIdx.x;
  const int wave = t >> 5;
  const int lane = t & 31;

  const int b  = blockIdx.z;
  const int cg = blockIdx.y;            // channel group (16 ch), 22 groups (pad 352)
  const int tr = blockIdx.x >> 3;       // patch-row of tile (8 rows)
  const int tc = blockIdx.x & 7;        // 32-col group
  const int r0 = tr * 8, c0 = tc * 32;
  const int cbase = cg * 16;

  // --- DCT-II orthonormal basis: M[i][j] ---
  if (t < 64) {
    int i = t >> 3, j = t & 7;
    lds_m[t] = (i == 0) ? 0.35355339059327378f
                        : 0.5f * cosf(3.14159265358979f * (float)i *
                                      (float)(2 * j + 1) * (1.0f / 16.0f));
  }
  // --- stage W_in chunk (16 x 64) as f16, zero rows past channel 339 ---
  for (int idx = t; idx < 16 * 64; idx += 256) {
    int r = idx >> 6, c = idx & 63;
    int ch = cbase + r;
    lds_w[idx] = (_Float16)((ch < C2) ? W_in[ch * 64 + c] : 0.0f);
  }
  // --- stage x tile via b128 loads: pixel-major, K-contiguous for B frags ---
  {
    const float* xb = x + (size_t)b * CIN * IMG2;
#pragma unroll 4
    for (int idx = t; idx < 64 * 64; idx += 256) {
      int c   = idx >> 6;               // input channel
      int rem = idx & 63;
      int row = rem >> 3, g4 = rem & 7; // 8 groups of 4 cols
      const float4 v = *(const float4*)&xb[(size_t)c * IMG2 +
                                           (size_t)(r0 + row) * IMG + c0 + g4 * 4];
      int pb = row * 32 + g4 * 4;
      lds_x[(pb + 0) * 64 + c] = (_Float16)v.x;
      lds_x[(pb + 1) * 64 + c] = (_Float16)v.y;
      lds_x[(pb + 2) * 64 + c] = (_Float16)v.z;
      lds_x[(pb + 3) * 64 + c] = (_Float16)v.w;
    }
  }
  __syncthreads();

  // --- GEMM: y[16 ch x 256 pix] = W_in * x,  K=64 as two f16 WMMA steps ---
  v8f acc0 = {}; v8f acc1 = {};
  const int nt0   = wave * 2;           // each wave owns 2 N-tiles (32 pixels)
  const int arow  = lane & 15;
  const int khalf = (lane >> 4) * 16;
#pragma unroll
  for (int s = 0; s < 2; ++s) {
    v16h a = *(const v16h*)&lds_w[arow * 64 + s * 32 + khalf];
    {
      int pix = nt0 * 16 + (lane & 15);
      v16h bf = *(const v16h*)&lds_x[pix * 64 + s * 32 + khalf];
      acc0 = __builtin_amdgcn_wmma_f32_16x16x32_f16(false, a, false, bf,
                                                    (short)0, acc0, false, false);
    }
    {
      int pix = (nt0 + 1) * 16 + (lane & 15);
      v16h bf = *(const v16h*)&lds_x[pix * 64 + s * 32 + khalf];
      acc1 = __builtin_amdgcn_wmma_f32_16x16x32_f16(false, a, false, bf,
                                                    (short)0, acc1, false, false);
    }
  }
  // scatter C fragments to lds_y[ch][pixel]  (VGPR r: lanes0-15 M=r, 16-31 M=8+r)
#pragma unroll
  for (int r = 0; r < 8; ++r) {
    int m = r + 8 * (lane >> 4);
    lds_y[m * 256 + nt0 * 16 + (lane & 15)]       = acc0[r];
    lds_y[m * 256 + (nt0 + 1) * 16 + (lane & 15)] = acc1[r];
  }
  __syncthreads();

  // --- DCT stage A:  D = quant ⊙ (M X M^T); 4 threads/patch, 2 rows each ---
  const int patch = t >> 2;             // 0..63 = 16 ch x 4 patch-cols
  const int chl   = patch >> 2;
  const int pc    = patch & 3;
  const int qrt   = t & 3;
  const int gch   = cbase + chl;
  const bool chValid = (gch < C2);
  float* X = lds_y + chl * 256 + pc * 8;   // element (p,q) at X[p*32+q]
  float* T = lds_t + chl * 256 + pc * 8;
#pragma unroll
  for (int ii = 0; ii < 2; ++ii) {
    int i = 2 * qrt + ii;
    float t1[8];
#pragma unroll
    for (int q = 0; q < 8; ++q) {
      float s = 0.f;
#pragma unroll
      for (int p = 0; p < 8; ++p) s = fmaf(lds_m[i * 8 + p], X[p * 32 + q], s);
      t1[q] = s;
    }
#pragma unroll
    for (int j = 0; j < 8; ++j) {
      float s = 0.f;
#pragma unroll
      for (int q = 0; q < 8; ++q) s = fmaf(t1[q], lds_m[j * 8 + q], s);
      float qv = chValid ? quant[(size_t)gch * 64 + i * 8 + j] : 1.0f;
      T[i * 32 + j] = s * qv;
    }
  }
  __syncthreads();

  // --- DCT stage B:  X' = M^T D M ---
#pragma unroll
  for (int ii = 0; ii < 2; ++ii) {
    int p = 2 * qrt + ii;
    float t3[8];
#pragma unroll
    for (int j = 0; j < 8; ++j) {
      float s = 0.f;
#pragma unroll
      for (int i = 0; i < 8; ++i) s = fmaf(lds_m[i * 8 + p], T[i * 32 + j], s);
      t3[j] = s;
    }
#pragma unroll
    for (int q = 0; q < 8; ++q) {
      float s = 0.f;
#pragma unroll
      for (int j = 0; j < 8; ++j) s = fmaf(t3[j], lds_m[j * 8 + q], s);
      X[p * 32 + q] = s;
    }
  }
  __syncthreads();

  // --- store yr tile as fp16 pairs (b32 stores, full-width cachelines) ---
  {
    _Float16* yrb = yr + (size_t)b * C2 * IMG2;
#pragma unroll 4
    for (int idx = t; idx < 16 * 128; idx += 256) { // 16 ch x 128 pixel-pairs
      int c  = idx >> 7;
      int pr = idx & 127;
      int gc = cbase + c;
      if (gc < C2) {
        HPack pk;
        pk.h[0] = (_Float16)lds_y[c * 256 + 2 * pr];
        pk.h[1] = (_Float16)lds_y[c * 256 + 2 * pr + 1];
        int row = pr >> 4, col = (2 * pr) & 31;
        *(unsigned*)&yrb[(size_t)gc * IMG2 + (size_t)(r0 + row) * IMG + c0 + col] = pk.u;
      }
    }
  }
}

// ---------------------------------------------------------------------------
// Kernel 2: depthwise 3x3 + exact GELU gate + project_out GEMM (f16 WMMA).
// Block = (8 rows x 32 cols) tile, all 64 out channels; K=170 padded to 192.
// ---------------------------------------------------------------------------
__global__ __launch_bounds__(256) void dctffn_k2(
    const _Float16* __restrict__ yr, const float* __restrict__ W_dw,
    const float* __restrict__ W_out, float* __restrict__ out)
{
  // LDS carve-up (64000 B):
  //   [0)      lds_wout f16[64*192]   24576 B  (W_out, K zero-padded)
  //   [24576)  lds_h    f16[32*10*36] 23040 B  (halo rows, origin gx = c0-2)
  //   [47616)  lds_g    f16[256*32]   16384 B  (gated activations, pixel-major)
  __shared__ __align__(32) unsigned char smem[64000];
  _Float16* lds_wout = (_Float16*)smem;
  _Float16* lds_h    = (_Float16*)(smem + 24576);
  _Float16* lds_g    = (_Float16*)(smem + 47616);

  const int t    = threadIdx.x;
  const int wave = t >> 5;
  const int lane = t & 31;

  const int b  = blockIdx.y;
  const int tr = blockIdx.x >> 3;
  const int tc = blockIdx.x & 7;
  const int r0 = tr * 8, c0 = tc * 32;

  // stage W_out (64 x 170 -> 64 x 192 f16, zero-padded K)
  for (int idx = t; idx < 64 * 192; idx += 256) {
    int r = idx / 192, k = idx - r * 192;
    lds_wout[idx] = (_Float16)((k < HID) ? W_out[r * HID + k] : 0.0f);
  }

  v8f acc[8] = {};                       // wave -> (mt, 8 N-tiles)
  const int mt  = wave & 3;
  const int ntb = (wave >> 2) * 8;

  const _Float16* yrb = yr + (size_t)b * C2 * IMG2;

  for (int kc = 0; kc < 6; ++kc) {       // K chunks of 32 (170 -> 192)
    const int j0 = kc * 32;
#pragma unroll 1
    for (int hph = 0; hph < 2; ++hph) {  // two 16-channel half-phases
      const int jb = j0 + hph * 16;
      // --- halo interior: gx in [c0, c0+32), packed u32 loads/stores ---
#pragma unroll 1
      for (int idx = t; idx < 32 * 10 * 16; idx += 256) {
        int s   = idx / 160;             // halo slot: 0-15 x1-ch, 16-31 x2-ch
        int rem = idx - s * 160;
        int hy  = rem >> 4, pp = rem & 15;
        int j   = jb + (s & 15);
        int ch  = (s < 16) ? j : j + HID;
        int gy  = r0 - 1 + hy;
        HPack pk; pk.u = 0u;
        if (j < HID && gy >= 0 && gy < IMG)
          pk.u = *(const unsigned*)&yrb[(size_t)ch * IMG2 + (size_t)gy * IMG + c0 + 2 * pp];
        *(unsigned*)&lds_h[(s * 10 + hy) * 36 + 2 + 2 * pp] = pk.u;
      }
      // --- halo edges: gx = c0-1 and gx = c0+32 (scalar b16) ---
      for (int idx = t; idx < 32 * 10 * 2; idx += 256) {
        int s   = idx / 20;
        int rem = idx - s * 20;
        int hy  = rem >> 1, e = rem & 1;
        int j   = jb + (s & 15);
        int ch  = (s < 16) ? j : j + HID;
        int gy  = r0 - 1 + hy;
        int gx  = e ? (c0 + 32) : (c0 - 1);
        float v = 0.0f;
        if (j < HID && gy >= 0 && gy < IMG && gx >= 0 && gx < IMG)
          v = (float)yrb[(size_t)ch * IMG2 + (size_t)gy * IMG + gx];
        lds_h[(s * 10 + hy) * 36 + (e ? 34 : 1)] = (_Float16)v;
      }
      __syncthreads();                   // also fences previous WMMA reads of lds_g
      // --- depthwise 3x3 + exact-GELU gate, 16 channels x 256 pixels ---
      {
        const int p = t, row = p >> 5, col = p & 31;
#pragma unroll 1
        for (int i = 0; i < 16; ++i) {
          int j = jb + i;
          float g = 0.0f;
          if (j < HID) {
            const float* w1 = W_dw + j * 9;
            const float* w2 = W_dw + (j + HID) * 9;
            float d1 = 0.f, d2 = 0.f;
#pragma unroll
            for (int ky = 0; ky < 3; ++ky)
#pragma unroll
              for (int kx = 0; kx < 3; ++kx) {
                // halo x-origin is gx = c0-2  ->  hx = col + kx + 1
                float h1 = (float)lds_h[(i * 10 + row + ky) * 36 + col + kx + 1];
                float h2 = (float)lds_h[((16 + i) * 10 + row + ky) * 36 + col + kx + 1];
                d1 = fmaf(w1[ky * 3 + kx], h1, d1);
                d2 = fmaf(w2[ky * 3 + kx], h2, d2);
              }
            g = 0.5f * d1 * (1.0f + erff(d1 * 0.70710678118f)) * d2;
          }
          lds_g[p * 32 + hph * 16 + i] = (_Float16)g;
        }
      }
      __syncthreads();
    }
    // --- WMMA accumulate this K-chunk: out += W_out[:, j0:j0+32] * g ---
    const int arow = 16 * mt + (lane & 15);
    const int kh   = (lane >> 4) * 16;
    v16h a = *(const v16h*)&lds_wout[arow * 192 + j0 + kh];
#pragma unroll
    for (int f = 0; f < 8; ++f) {
      int pix = (ntb + f) * 16 + (lane & 15);
      v16h bf = *(const v16h*)&lds_g[pix * 32 + kh];
      acc[f] = __builtin_amdgcn_wmma_f32_16x16x32_f16(false, a, false, bf,
                                                      (short)0, acc[f], false, false);
    }
    __syncthreads();
  }

  // store 64 x 256 fp32 tile (lanes 0-15 write 16 consecutive cols -> coalesced)
  float* ob = out + (size_t)b * 64 * IMG2;
#pragma unroll
  for (int f = 0; f < 8; ++f) {
    int n = (ntb + f) * 16 + (lane & 15);
    int row = n >> 5, col = n & 31;
    size_t gbase = (size_t)(r0 + row) * IMG + (c0 + col);
#pragma unroll
    for (int r = 0; r < 8; ++r) {
      int ch = 16 * mt + r + 8 * (lane >> 4);
      ob[(size_t)ch * IMG2 + gbase] = acc[f][r];
    }
  }
}

// ---------------------------------------------------------------------------
extern "C" void kernel_launch(void* const* d_in, const int* in_sizes, int n_in,
                              void* d_out, int out_size, void* d_ws, size_t ws_size,
                              hipStream_t stream) {
  (void)in_sizes; (void)n_in; (void)out_size; (void)ws_size;
  const float* x     = (const float*)d_in[0];
  const float* W_in  = (const float*)d_in[1];
  const float* W_dw  = (const float*)d_in[2];
  const float* quant = (const float*)d_in[3];
  const float* W_out = (const float*)d_in[4];
  float*       outp  = (float*)d_out;
  _Float16*    yr    = (_Float16*)d_ws;   // 4*340*256*256 halves = 178 MB

  dim3 g1(256, 22, 4);                    // 256 tiles x 22 ch-groups x 4 batch
  dctffn_k1<<<g1, 256, 0, stream>>>(x, W_in, quant, yr);
  dim3 g2(256, 4);                        // 256 tiles x 4 batch
  dctffn_k2<<<g2, 256, 0, stream>>>(yr, W_dw, W_out, outp);
}
